// ModelInverse_44942537785516
// MI455X (gfx1250) — compile-verified, matrix operand order checked by reading, not run on hardware
//
#include <hip/hip_runtime.h>

#define HDIM   64
#define MONO_C 1e-3f
#define N_ITER 20

typedef __attribute__((ext_vector_type(16))) _Float16 v16h;
typedef __attribute__((ext_vector_type(8)))  float    v8f;

__device__ __forceinline__ float sigf(float x) {
  // 1 / (1 + e^-x) via v_exp_f32 + v_rcp_f32
  return __builtin_amdgcn_rcpf(1.0f + __expf(-x));
}

// ---------------------------------------------------------------------------
// Setup: exponentiate weights once; compute A(0), A(1).
// ws layout (float idx): [0..63]=exp(w1), [64..127]=exp(w3), [128]=a0, [129]=a1
//                        bytes 1024.. : exp(w2) as f16, row-major [n][k] (64x64)
// ---------------------------------------------------------------------------
__global__ __launch_bounds__(64) void mi_setup(
    const float* __restrict__ pw1, const float* __restrict__ b1,
    const float* __restrict__ pw2, const float* __restrict__ b2,
    const float* __restrict__ pw3, const float* __restrict__ b3,
    float* __restrict__ wsf) {
  __shared__ float ew1[HDIM], ew3[HDIM], ew2[HDIM * HDIM], h1[HDIM], h2[HDIM];
  const int j = threadIdx.x;
  _Float16* wsh = (_Float16*)(wsf + 256);

  float e1 = __expf(pw1[j]); ew1[j] = e1; wsf[j] = e1;
  float e3 = __expf(pw3[j]); ew3[j] = e3; wsf[64 + j] = e3;
  for (int k = 0; k < HDIM; ++k) {
    float e2 = __expf(pw2[j * HDIM + k]);
    ew2[j * HDIM + k] = e2;
    wsh[j * HDIM + k] = (_Float16)e2;
  }
  __syncthreads();

  for (int xi = 0; xi < 2; ++xi) {
    float x = (float)xi;
    h1[j] = sigf(fmaf(x, ew1[j], b1[j]));
    __syncthreads();
    float s = b2[j];
    for (int k = 0; k < HDIM; ++k) s = fmaf(ew2[j * HDIM + k], h1[k], s);
    h2[j] = sigf(s);
    __syncthreads();
    if (j == 0) {
      float s3 = b3[0];
      for (int k = 0; k < HDIM; ++k) s3 = fmaf(ew3[k], h2[k], s3);
      wsf[128 + xi] = sigf(s3) + MONO_C * x;   // a0 / a1
    }
    __syncthreads();
  }
}

// ---------------------------------------------------------------------------
// Main: 8 waves/block, 16 samples/wave; hidden layer via 8x wmma_f32_16x16x32_f16
// ---------------------------------------------------------------------------
__global__ __launch_bounds__(256) void mi_invert(
    const float* __restrict__ z,  const float* __restrict__ b1,
    const float* __restrict__ b2, const float* __restrict__ b3,
    const float* __restrict__ wsf, float* __restrict__ out, int n) {
  __shared__ float    s_ew1[HDIM], s_b1[HDIM], s_b2[HDIM], s_ew3[HDIM];
  __shared__ _Float16 s_ew2[HDIM * HDIM];
  __shared__ float    s_y[8][16];

  const int tid  = threadIdx.x;
  const int wave = tid >> 5, lane = tid & 31;
  const int ln   = lane & 15, half = lane >> 4;

  if (tid < HDIM) {
    s_ew1[tid] = wsf[tid];
    s_ew3[tid] = wsf[64 + tid];
    s_b1[tid]  = b1[tid];
    s_b2[tid]  = b2[tid];
  }
  const _Float16* wsh = (const _Float16*)(wsf + 256);
#pragma unroll
  for (int t = 0; t < 16; ++t) s_ew2[tid + t * 256] = wsh[tid + t * 256];
  __syncthreads();

  // B fragments: exp(W2)^T in WMMA 16-bit B layout, 4 N-tiles x 2 K-halves.
  // element e of lane (16*half+ln): k = kh*32 + (e<8 ? half*8+e : 16+half*8+e-8)
  v16h Bf[4][2];
#pragma unroll
  for (int g = 0; g < 4; ++g)
#pragma unroll
    for (int kh = 0; kh < 2; ++kh)
#pragma unroll
      for (int e = 0; e < 16; ++e) {
        int k = kh * 32 + ((e < 8) ? (half * 8 + e) : (16 + half * 8 + (e - 8)));
        Bf[g][kh][e] = s_ew2[(g * 16 + ln) * HDIM + k];
      }

  // Per-lane layer-1 constants in A-fragment order (32 of 64 rows per lane).
  float rew1[32], rb1[32];
#pragma unroll
  for (int e = 0; e < 16; ++e) {
    int k0 = (e < 8) ? (half * 8 + e) : (16 + half * 8 + (e - 8));
    rew1[e]      = s_ew1[k0];      rb1[e]      = s_b1[k0];
    rew1[16 + e] = s_ew1[k0 + 32]; rb1[16 + e] = s_b1[k0 + 32];
  }
  float rb2[4], rew3[4];
#pragma unroll
  for (int g = 0; g < 4; ++g) {
    rb2[g]  = s_b2[g * 16 + ln];
    rew3[g] = s_ew3[g * 16 + ln];
  }

  // Per-sample state, replicated in both 16-lane halves.
  const int sidx = blockIdx.x * 128 + wave * 16 + ln;
  const int cidx = (sidx < n) ? sidx : (n - 1);
  const float a0 = wsf[128], a1 = wsf[129];
  const float tgt = fmaf(z[cidx], a1 - a0, a0);   // F(mid)>z  <=>  A(mid)>tgt
  const float b3v = b3[0];

  float lo = 0.0f, hi = 1.0f;
#pragma unroll 1
  for (int it = 0; it < N_ITER; ++it) {
    const float mid = 0.5f * (lo + hi);

    // Layer 1: sigmoid(mid*exp(w1)+b1), generated directly as A fragments.
    v16h A0, A1;
#pragma unroll
    for (int e = 0; e < 16; ++e) {
      A0[e] = (_Float16)sigf(fmaf(mid, rew1[e],      rb1[e]));
      A1[e] = (_Float16)sigf(fmaf(mid, rew1[16 + e], rb1[16 + e]));
    }

    // Layer 2 (16x64 @ 64x64) + layer-3 weighting.
    float acc[8] = {0.f, 0.f, 0.f, 0.f, 0.f, 0.f, 0.f, 0.f};
#pragma unroll
    for (int g = 0; g < 4; ++g) {
      v8f c = {};
      c = __builtin_amdgcn_wmma_f32_16x16x32_f16(false, A0, false, Bf[g][0],
                                                 (short)0, c, false, false);
      c = __builtin_amdgcn_wmma_f32_16x16x32_f16(false, A1, false, Bf[g][1],
                                                 (short)0, c, false, false);
#pragma unroll
      for (int i = 0; i < 8; ++i)
        acc[i] = fmaf(sigf(c[i] + rb2[g]), rew3[g], acc[i]);
    }

    // Row-sum across each 16-lane half (C row m=i in lanes 0-15, m=i+8 in 16-31).
#pragma unroll
    for (int i = 0; i < 8; ++i) {
      float v = acc[i];
      v += __shfl_xor(v, 1, 32);
      v += __shfl_xor(v, 2, 32);
      v += __shfl_xor(v, 4, 32);
      v += __shfl_xor(v, 8, 32);
      if (ln == i) s_y[wave][half * 8 + i] = v;   // same-wave LDS: in-order
    }
    const float Fm = sigf(s_y[wave][ln] + b3v) + MONO_C * mid;

    const bool gt = Fm > tgt;
    hi = gt ? mid : hi;
    lo = gt ? lo  : mid;
  }

  if (half == 0 && sidx < n) out[sidx] = 0.5f * (lo + hi);
}

// ---------------------------------------------------------------------------
extern "C" void kernel_launch(void* const* d_in, const int* in_sizes, int n_in,
                              void* d_out, int out_size, void* d_ws, size_t ws_size,
                              hipStream_t stream) {
  const float* z   = (const float*)d_in[0];
  const float* pw1 = (const float*)d_in[1];
  const float* b1  = (const float*)d_in[2];
  const float* pw2 = (const float*)d_in[3];
  const float* b2  = (const float*)d_in[4];
  const float* pw3 = (const float*)d_in[5];
  const float* b3  = (const float*)d_in[6];
  float* out = (float*)d_out;
  float* wsf = (float*)d_ws;   // needs 1024 B f32 consts + 8192 B f16 W2
  const int n = in_sizes[0];

  mi_setup<<<1, 64, 0, stream>>>(pw1, b1, pw2, b2, pw3, b3, wsf);
  const int blocks = (n + 127) / 128;   // 16 samples/wave * 8 waves
  mi_invert<<<blocks, 256, 0, stream>>>(z, b1, b2, b3, wsf, out, n);
}
